// Home_temp_enocder_23313082483273
// MI455X (gfx1250) — matrix operand order repaired and checked by reading, not compiled, not run on hardware
//
#include <hip/hip_runtime.h>

// ---------------------------------------------------------------------------
// Problem constants (match the JAX reference)
// ---------------------------------------------------------------------------
#define NROWS 65536
#define CIN   8
#define COUT  128
#define HDIM  128
#define OUTD  256
#define TT    5
#define EPSV  1e-5f
#define SLOPE 0.05f

typedef __attribute__((ext_vector_type(16))) _Float16 v16h;
typedef __attribute__((ext_vector_type(2)))  _Float16 hpair;
typedef __attribute__((ext_vector_type(8)))  float    v8f;

// ---------------------------------------------------------------------------
// WMMA fragment loaders (contiguous k = 16*khalf + s in A and B, 16-bit 16x16x32).
// Each compiles to two global_load_b128 per fragment (verified in disasm).
// ---------------------------------------------------------------------------
static __device__ __forceinline__ v16h load_a_frag(const _Float16* __restrict__ Arow,
                                                   int k0, int khalf) {
  v16h a;
#pragma unroll
  for (int g = 0; g < 8; ++g) {
    const int kb = 2 * g + 16 * khalf;
    hpair p = *reinterpret_cast<const hpair*>(Arow + k0 + kb);
    a[2 * g] = p[0];
    a[2 * g + 1] = p[1];
  }
  return a;
}

static __device__ __forceinline__ v16h load_b_frag(const _Float16* __restrict__ p) {
  v16h b;
#pragma unroll
  for (int v = 0; v < 8; ++v) {
    hpair q = *reinterpret_cast<const hpair*>(p + 2 * v);
    b[2 * v] = q[0];
    b[2 * v + 1] = q[1];
  }
  return b;
}

// One K=128 GEMM segment: acc[0..2] += A(16x128 rows row0..row0+15) @ W^T for
// this wave's 3 column tiles.  A and W are direct kernarg-derived pointers so
// the loads lower to global_load_b128 (not flat_load).
static __device__ __forceinline__ void gemm_k128(
    v8f acc[3], const _Float16* __restrict__ A,
    const _Float16* __restrict__ W, int ldw, int kofs,
    int row0, int m, int khalf, int wave) {
  const _Float16* __restrict__ Arow = A + (size_t)(row0 + m) * HDIM;
#pragma unroll
  for (int ks = 0; ks < HDIM / 32; ++ks) {
    v16h a = load_a_frag(Arow, ks * 32, khalf);
#pragma unroll
    for (int ct = 0; ct < 3; ++ct) {
      const int col = (wave * 3 + ct) * 16 + m;
      v16h b = load_b_frag(W + (size_t)col * ldw + kofs + ks * 32 + 16 * khalf);
      acc[ct] = __builtin_amdgcn_wmma_f32_16x16x32_f16(
          false, a, false, b, (short)0, acc[ct], false, false);
    }
  }
}

// ---------------------------------------------------------------------------
// fp32 -> fp16 weight conversion
// ---------------------------------------------------------------------------
__global__ void f32_to_f16_kernel(const float* __restrict__ src,
                                  _Float16* __restrict__ dst, int n) {
  int i = blockIdx.x * blockDim.x + threadIdx.x;
  if (i < n) dst[i] = (_Float16)src[i];
}

// ---------------------------------------------------------------------------
// Conv1d(k=3, pad=1, no bias) over [N,CIN,T].  mode 0: accumulate per-channel
// sum/sumsq (BN training stats).  mode 1: recompute conv (24 FMA/out), apply
// BN affine + ReLU, emit fp16 time-major x16[t][n][c].
// ---------------------------------------------------------------------------
__global__ __launch_bounds__(128) void conv_kernel(
    const float* __restrict__ tracks, const float* __restrict__ conv_w,
    float* __restrict__ gsum, float* __restrict__ gsumsq,
    const float* __restrict__ scale, const float* __restrict__ shift,
    _Float16* __restrict__ x16, int mode, int n_per_block) {
  const int c = threadIdx.x;
  float w[CIN][3];
#pragma unroll
  for (int ci = 0; ci < CIN; ++ci)
#pragma unroll
    for (int kk = 0; kk < 3; ++kk)
      w[ci][kk] = conv_w[((size_t)c * CIN + ci) * 3 + kk];

  float sc = 0.f, sh = 0.f;
  if (mode == 1) { sc = scale[c]; sh = shift[c]; }

  __shared__ float tr[CIN][TT + 2];  // zero-padded window
  float s = 0.f, sq = 0.f;
  const int n0 = blockIdx.x * n_per_block;
  for (int n = n0; n < n0 + n_per_block; ++n) {
    __syncthreads();
    if (threadIdx.x < CIN * TT) {
      int ci = threadIdx.x / TT, t = threadIdx.x % TT;
      tr[ci][t + 1] = tracks[((size_t)n * CIN + ci) * TT + t];
    } else if (threadIdx.x < CIN * TT + 2 * CIN) {
      int idx = threadIdx.x - CIN * TT;
      tr[idx >> 1][(idx & 1) ? (TT + 1) : 0] = 0.f;
    }
    __syncthreads();
#pragma unroll
    for (int t = 0; t < TT; ++t) {
      float val = 0.f;
#pragma unroll
      for (int ci = 0; ci < CIN; ++ci)
#pragma unroll
        for (int kk = 0; kk < 3; ++kk)
          val += tr[ci][t + kk] * w[ci][kk];
      if (mode == 0) {
        s += val;
        sq += val * val;
      } else {
        float y = val * sc + sh;
        y = y > 0.f ? y : 0.f;  // ReLU
        x16[((size_t)t * NROWS + n) * COUT + c] = (_Float16)y;
      }
    }
  }
  if (mode == 0) {
    atomicAdd(&gsum[c], s);
    atomicAdd(&gsumsq[c], sq);
  }
}

__global__ void finalize_kernel(const float* __restrict__ gsum,
                                const float* __restrict__ gsumsq,
                                const float* __restrict__ gamma,
                                const float* __restrict__ beta,
                                float* __restrict__ scale,
                                float* __restrict__ shift) {
  int c = threadIdx.x;
  const float inv = 1.f / ((float)NROWS * (float)TT);
  float mean = gsum[c] * inv;
  float var = gsumsq[c] * inv - mean * mean;
  float sc = gamma[c] * rsqrtf(var + EPSV);
  scale[c] = sc;
  shift[c] = beta[c] - mean * sc;
}

// ---------------------------------------------------------------------------
// One GRU timestep for 16 rows per workgroup (8 wave32 waves).
// gi = [xA|xB] @ Wih^T  (K = 128 or 256),  gh = h @ Whh^T  (K = 128).
// 24 gi + 24 gh column tiles of 16 -> 3+3 WMMA accumulators per wave.
// Gate tiles land in LDS, then the fused GRU cell (PyTorch math) runs.
// ---------------------------------------------------------------------------
__global__ __launch_bounds__(256) void gru_step_kernel(
    const _Float16* __restrict__ xA, const _Float16* __restrict__ xB,
    const _Float16* __restrict__ Wih, int has_xB,
    const _Float16* __restrict__ Whh,
    const float* __restrict__ bih, const float* __restrict__ bhh,
    const _Float16* __restrict__ h_old, _Float16* __restrict__ h_new,
    _Float16* __restrict__ seq_out) {
  __shared__ float ldsG[2][16][3 * HDIM + 8];  // [gi|gh][row][col] ~49KB
  const int row0 = blockIdx.x * 16;
  const int wave = threadIdx.x >> 5;
  const int lane = threadIdx.x & 31;
  const int m = lane & 15;      // row (A) / column (B, C/D) lane index
  const int khalf = lane >> 4;  // K half-select
  const int ldwih = has_xB ? (COUT + HDIM) : HDIM;

  // ---- gi = x @ Wih^T ----
  v8f accGi[3] = {};
  gemm_k128(accGi, xA, Wih, ldwih, 0, row0, m, khalf, wave);
  if (has_xB)
    gemm_k128(accGi, xB, Wih, ldwih, HDIM, row0, m, khalf, wave);

  // ---- gh = h @ Whh^T ----
  v8f accGh[3] = {};
  gemm_k128(accGh, h_old, Whh, HDIM, 0, row0, m, khalf, wave);

  // C/D layout: VGPR i -> row i + 8*khalf, column = coltile*16 + m
#pragma unroll
  for (int ct = 0; ct < 3; ++ct) {
    const int col = (wave * 3 + ct) * 16 + m;
#pragma unroll
    for (int i = 0; i < 8; ++i) {
      ldsG[0][i + 8 * khalf][col] = accGi[ct][i];
      ldsG[1][i + 8 * khalf][col] = accGh[ct][i];
    }
  }
  __syncthreads();

  // ---- fused GRU cell: r,z,n gates + state update ----
  for (int idx = threadIdx.x; idx < 16 * HDIM; idx += 256) {
    const int mm = idx >> 7;
    const int j = idx & (HDIM - 1);
    float gr = ldsG[0][mm][j] + bih[j];
    float hr = ldsG[1][mm][j] + bhh[j];
    float gz = ldsG[0][mm][HDIM + j] + bih[HDIM + j];
    float hz = ldsG[1][mm][HDIM + j] + bhh[HDIM + j];
    float gn = ldsG[0][mm][2 * HDIM + j] + bih[2 * HDIM + j];
    float hn = ldsG[1][mm][2 * HDIM + j] + bhh[2 * HDIM + j];
    float r = 1.f / (1.f + __expf(-(gr + hr)));
    float z = 1.f / (1.f + __expf(-(gz + hz)));
    float nn = tanhf(gn + r * hn);
    float hv = (1.f - z) * nn + z * (float)h_old[(size_t)(row0 + mm) * HDIM + j];
    _Float16 hh = (_Float16)hv;
    h_new[(size_t)(row0 + mm) * HDIM + j] = hh;
    seq_out[(size_t)(row0 + mm) * HDIM + j] = hh;
  }
}

// ---------------------------------------------------------------------------
// Decoder: flat[n][t*H+c] = feats16[t][n][c]; y = flat @ Wdec^T + b (WMMA,
// K=640 in 20 steps), then per-row LayerNorm (16 threads/row) + leaky ReLU.
// ---------------------------------------------------------------------------
__global__ __launch_bounds__(256) void decoder_kernel(
    const _Float16* __restrict__ feats,  // [T][N][HDIM] fp16
    const _Float16* __restrict__ Wdec,   // [OUTD][T*HDIM] fp16
    const float* __restrict__ dec_b, const float* __restrict__ ln_w,
    const float* __restrict__ ln_b, float* __restrict__ out) {
  __shared__ float ldsY[16][OUTD + 8];
  __shared__ float partS[16][16], partQ[16][16];
  __shared__ float rMean[16], rInv[16];
  const int row0 = blockIdx.x * 16;
  const int wave = threadIdx.x >> 5;
  const int lane = threadIdx.x & 31;
  const int m = lane & 15;
  const int khalf = lane >> 4;

  v8f acc[2] = {};
  for (int ks = 0; ks < (TT * HDIM) / 32; ++ks) {
    const int k0 = ks * 32;
    const int t = k0 >> 7;           // timestep segment of the flat K axis
    const int c0 = k0 & (HDIM - 1);  // offset within segment (32-aligned)
    const _Float16* __restrict__ Arow = feats + ((size_t)t * NROWS + row0 + m) * HDIM;
    v16h a = load_a_frag(Arow, c0, khalf);
#pragma unroll
    for (int cc = 0; cc < 2; ++cc) {
      const int col = (wave * 2 + cc) * 16 + m;
      v16h b = load_b_frag(Wdec + (size_t)col * (TT * HDIM) + k0 + 16 * khalf);
      acc[cc] = __builtin_amdgcn_wmma_f32_16x16x32_f16(
          false, a, false, b, (short)0, acc[cc], false, false);
    }
  }
#pragma unroll
  for (int cc = 0; cc < 2; ++cc) {
    const int col = (wave * 2 + cc) * 16 + m;
#pragma unroll
    for (int i = 0; i < 8; ++i)
      ldsY[i + 8 * khalf][col] = acc[cc][i] + dec_b[col];
  }
  __syncthreads();

  // LayerNorm stats: 16 threads per row, 16-wide strided partials
  {
    const int rrow = threadIdx.x >> 4;
    const int part = threadIdx.x & 15;
    float s = 0.f, sq = 0.f;
#pragma unroll
    for (int j = part; j < OUTD; j += 16) {
      float v = ldsY[rrow][j];
      s += v;
      sq += v * v;
    }
    partS[rrow][part] = s;
    partQ[rrow][part] = sq;
  }
  __syncthreads();
  if (threadIdx.x < 16) {
    float s = 0.f, sq = 0.f;
#pragma unroll
    for (int p = 0; p < 16; ++p) {
      s += partS[threadIdx.x][p];
      sq += partQ[threadIdx.x][p];
    }
    float mean = s * (1.f / OUTD);
    float var = sq * (1.f / OUTD) - mean * mean;
    rMean[threadIdx.x] = mean;
    rInv[threadIdx.x] = rsqrtf(var + EPSV);
  }
  __syncthreads();

  for (int idx = threadIdx.x; idx < 16 * OUTD; idx += 256) {
    const int mm = idx >> 8;
    const int j = idx & (OUTD - 1);
    float v = (ldsY[mm][j] - rMean[mm]) * rInv[mm] * ln_w[j] + ln_b[j];
    out[(size_t)(row0 + mm) * OUTD + j] = (v >= 0.f) ? v : SLOPE * v;
  }
}

// ---------------------------------------------------------------------------
// Host orchestration (all on `stream`; graph-capture safe: no sync/alloc)
// ---------------------------------------------------------------------------
extern "C" void kernel_launch(void* const* d_in, const int* in_sizes, int n_in,
                              void* d_out, int out_size, void* d_ws, size_t ws_size,
                              hipStream_t stream) {
  (void)in_sizes; (void)n_in; (void)out_size; (void)ws_size;

  const float* tracks   = (const float*)d_in[0];
  const float* conv_w   = (const float*)d_in[1];
  const float* bn_gamma = (const float*)d_in[2];
  const float* bn_beta  = (const float*)d_in[3];
  const float* g1_wih   = (const float*)d_in[4];
  const float* g1_whh   = (const float*)d_in[5];
  const float* g1_bih   = (const float*)d_in[6];
  const float* g1_bhh   = (const float*)d_in[7];
  const float* g2_wih   = (const float*)d_in[8];
  const float* g2_whh   = (const float*)d_in[9];
  const float* g2_bih   = (const float*)d_in[10];
  const float* g2_bhh   = (const float*)d_in[11];
  const float* dec_w    = (const float*)d_in[12];
  const float* dec_b    = (const float*)d_in[13];
  const float* ln_w     = (const float*)d_in[14];
  const float* ln_b     = (const float*)d_in[15];

  char* ws = (char*)d_ws;
  size_t off = 0;
  auto take = [&](size_t bytes) -> char* {
    char* p = ws + off;
    off = (off + bytes + 255) & ~(size_t)255;
    return p;
  };

  _Float16* wih1 = (_Float16*)take((size_t)3 * HDIM * COUT * 2);          // [384][128]
  _Float16* whh1 = (_Float16*)take((size_t)3 * HDIM * HDIM * 2);          // [384][128]
  _Float16* wih2 = (_Float16*)take((size_t)3 * HDIM * (COUT + HDIM) * 2); // [384][256]
  _Float16* whh2 = (_Float16*)take((size_t)3 * HDIM * HDIM * 2);          // [384][128]
  _Float16* wdec = (_Float16*)take((size_t)OUTD * TT * HDIM * 2);         // [256][640]
  float* gsum   = (float*)take(2 * COUT * sizeof(float));                 // sum | sumsq
  float* gsumsq = gsum + COUT;
  float* scale  = (float*)take(2 * COUT * sizeof(float));                 // scale | shift
  float* shift  = scale + COUT;
  _Float16* x16    = (_Float16*)take((size_t)TT * NROWS * COUT * 2);
  _Float16* hs16   = (_Float16*)take((size_t)TT * NROWS * HDIM * 2);
  _Float16* feat16 = (_Float16*)take((size_t)TT * NROWS * HDIM * 2);
  _Float16* h16a   = (_Float16*)take((size_t)NROWS * HDIM * 2);
  _Float16* h16b   = (_Float16*)take((size_t)NROWS * HDIM * 2);

  // Weight conversion to fp16 (tiny; weights then live in L2)
  auto cvt = [&](const float* s, _Float16* d, int n) {
    f32_to_f16_kernel<<<(n + 255) / 256, 256, 0, stream>>>(s, d, n);
  };
  cvt(g1_wih, wih1, 3 * HDIM * COUT);
  cvt(g1_whh, whh1, 3 * HDIM * HDIM);
  cvt(g2_wih, wih2, 3 * HDIM * (COUT + HDIM));
  cvt(g2_whh, whh2, 3 * HDIM * HDIM);
  cvt(dec_w,  wdec, OUTD * TT * HDIM);

  // Conv + BN(training stats) + ReLU -> x16 [T][N][128] fp16
  hipMemsetAsync(gsum, 0, 2 * COUT * sizeof(float), stream);
  conv_kernel<<<1024, 128, 0, stream>>>(tracks, conv_w, gsum, gsumsq,
                                        nullptr, nullptr, nullptr, 0, NROWS / 1024);
  finalize_kernel<<<1, 128, 0, stream>>>(gsum, gsumsq, bn_gamma, bn_beta, scale, shift);
  conv_kernel<<<1024, 128, 0, stream>>>(tracks, conv_w, gsum, gsumsq,
                                        scale, shift, x16, 1, NROWS / 1024);

  // GRU1: reverse scan; per-step WMMA GEMM + fused cell
  hipMemsetAsync(h16a, 0, (size_t)NROWS * HDIM * 2, stream);
  _Float16* hold = h16a;
  _Float16* hnew = h16b;
  for (int step = 0; step < TT; ++step) {
    const int t = TT - 1 - step;
    gru_step_kernel<<<NROWS / 16, 256, 0, stream>>>(
        x16 + (size_t)t * NROWS * HDIM, nullptr, wih1, 0, whh1,
        g1_bih, g1_bhh, hold, hnew, hs16 + (size_t)t * NROWS * HDIM);
    _Float16* tmp = hold; hold = hnew; hnew = tmp;
  }

  // GRU2: forward scan over aug = [x | hidden_states] (two K=128 segments)
  hipMemsetAsync(h16a, 0, (size_t)NROWS * HDIM * 2, stream);
  hold = h16a;
  hnew = h16b;
  for (int t = 0; t < TT; ++t) {
    gru_step_kernel<<<NROWS / 16, 256, 0, stream>>>(
        x16 + (size_t)t * NROWS * HDIM, hs16 + (size_t)t * NROWS * HDIM,
        wih2, 1, whh2, g2_bih, g2_bhh, hold, hnew,
        feat16 + (size_t)t * NROWS * HDIM);
    _Float16* tmp = hold; hold = hnew; hnew = tmp;
  }

  // Decoder GEMM (K=640) + LayerNorm + leaky ReLU -> fp32 output
  decoder_kernel<<<NROWS / 16, 256, 0, stream>>>(feat16, wdec, dec_b, ln_w, ln_b,
                                                 (float*)d_out);
}
